// Block_42382737277039
// MI455X (gfx1250) — compile-verified, matrix operand order checked by reading, not run on hardware
//
#include <hip/hip_runtime.h>
#include <hip/hip_bf16.h>

// ---------------------------------------------------------------------------
// Feature detection: CDNA5 async global->LDS DMA (ASYNCcnt-tracked)
// ---------------------------------------------------------------------------
#ifndef __has_builtin
#define __has_builtin(x) 0
#endif
#if __has_builtin(__builtin_amdgcn_global_load_async_to_lds_b128) && \
    __has_builtin(__builtin_amdgcn_s_wait_asynccnt)
#define USE_ASYNC_LDS 1
#else
#define USE_ASYNC_LDS 0
#endif

// ---------------------------------------------------------------------------
// Types
// ---------------------------------------------------------------------------
typedef __bf16 bf16;
typedef __attribute__((ext_vector_type(16))) __bf16 bf16x16;
typedef __attribute__((ext_vector_type(8)))  float  f32x8;
typedef __attribute__((ext_vector_type(4)))  int    v4i;

union FragAB { uint4 q[2]; bf16x16 v; };   // 32B: one WMMA A/B fragment (16 bf16)

__device__ inline bf16 to_bf16(float f) {
  union { float f; unsigned u; } a; a.f = f;
  unsigned r = a.u + 0x7FFFu + ((a.u >> 16) & 1u);   // round-to-nearest-even
  union { unsigned short s; bf16 b; } o; o.s = (unsigned short)(r >> 16);
  return o.b;
}

__device__ inline float gelu_tanh(float x) {
  float x3 = x * x * x;
  float u = 0.7978845608028654f * (x + 0.044715f * x3);
  return 0.5f * x * (1.0f + tanhf(u));
}

// Copy 16 bytes (8 bf16) global -> LDS. Async DMA on CDNA5, sync fallback.
// Builtin signature (per compiler diagnostic): (AS1 v4i32*, AS3 v4i32*, imm, imm)
__device__ inline void cp16B_g2l(const bf16* g, bf16* l) {
#if USE_ASYNC_LDS
  __builtin_amdgcn_global_load_async_to_lds_b128(
      (__attribute__((address_space(1))) v4i*)g,
      (__attribute__((address_space(3))) v4i*)l, 0, 0);
#else
  *(uint4*)l = *(const uint4*)g;
#endif
}

__device__ inline void wait_async_all() {
#if USE_ASYNC_LDS
  __builtin_amdgcn_s_wait_asynccnt(0);
#endif
}

// ---------------------------------------------------------------------------
// fp32 [K][N] -> bf16 [N][K] transposed weights (tiled through LDS).
// Done once per call so every GEMM B-tile is a contiguous row-major load.
// ---------------------------------------------------------------------------
__global__ __launch_bounds__(256)
void cvt_transpose_f32_bf16(const float* __restrict__ in, bf16* __restrict__ out,
                            int K, int N) {
  __shared__ float tile[32][33];
  const int k0 = blockIdx.y * 32, n0 = blockIdx.x * 32;
  const int tx = threadIdx.x & 31, ty = threadIdx.x >> 5;
#pragma unroll
  for (int i = ty; i < 32; i += 8)
    tile[i][tx] = in[(size_t)(k0 + i) * N + (n0 + tx)];
  __syncthreads();
#pragma unroll
  for (int i = ty; i < 32; i += 8)
    out[(size_t)(n0 + i) * K + (k0 + tx)] = to_bf16(tile[tx][i]);
}

// ---------------------------------------------------------------------------
// V pre-transpose: qkv[b, s, 2D + h*64 + d] -> vt[(b*16+h)*64 + d][s]
// One 8MB pass so the flash-attention inner loop stages V^T tiles with pure
// contiguous async-DMA copies (no scalar LDS transposes per key tile).
// ---------------------------------------------------------------------------
__global__ __launch_bounds__(256)
void transpose_v_kernel(const bf16* __restrict__ qkv, bf16* __restrict__ vt, int S) {
  __shared__ bf16 tile[32][33];
  const int s0 = blockIdx.x * 32;
  const int h  = blockIdx.y >> 1, d0 = (blockIdx.y & 1) * 32;
  const int b  = blockIdx.z;
  const int tx = threadIdx.x & 31, ty = threadIdx.x >> 5;
  const bf16* src = qkv + ((size_t)b * S) * 3072 + 2048 + h * 64 + d0;
#pragma unroll
  for (int i = ty; i < 32; i += 8)
    tile[i][tx] = src[(size_t)(s0 + i) * 3072 + tx];
  __syncthreads();
  bf16* dst = vt + ((size_t)(b * 16 + h) * 64 + d0) * S;
#pragma unroll
  for (int i = ty; i < 32; i += 8)
    dst[(size_t)i * S + s0 + tx] = tile[tx][i];
}

// ---------------------------------------------------------------------------
// LayerNorm over D=1024, writes bf16 activations
// ---------------------------------------------------------------------------
__global__ __launch_bounds__(256)
void ln_bf16_kernel(const float* __restrict__ x, const float* __restrict__ g,
                    const float* __restrict__ b, bf16* __restrict__ out) {
  const int row = blockIdx.x;
  const int t = threadIdx.x;
  const float* xr = x + (size_t)row * 1024;
  float v[4], s = 0.f, ss = 0.f;
#pragma unroll
  for (int i = 0; i < 4; ++i) {
    v[i] = xr[t + i * 256];
    s += v[i]; ss += v[i] * v[i];
  }
#pragma unroll
  for (int off = 16; off >= 1; off >>= 1) {
    s  += __shfl_xor(s,  off, 32);
    ss += __shfl_xor(ss, off, 32);
  }
  __shared__ float red[2][8];
  if ((t & 31) == 0) { red[0][t >> 5] = s; red[1][t >> 5] = ss; }
  __syncthreads();
  float ts = 0.f, tss = 0.f;
#pragma unroll
  for (int w = 0; w < 8; ++w) { ts += red[0][w]; tss += red[1][w]; }
  const float mu  = ts * (1.0f / 1024.0f);
  const float var = tss * (1.0f / 1024.0f) - mu * mu;
  const float inv = rsqrtf(var + 1e-5f);
#pragma unroll
  for (int i = 0; i < 4; ++i) {
    int c = t + i * 256;
    out[(size_t)row * 1024 + c] = to_bf16((v[i] - mu) * inv * g[c] + b[c]);
  }
}

// ---------------------------------------------------------------------------
// Tiled WMMA bf16 GEMM: C = A[MxK] * Bt[NxK]^T (+bias, +epilogue)
// Block tile 128x128, BK=32, 8 waves (2x4), wave tile 64x32 (4x2 WMMA tiles).
// Double-buffered LDS, async global->LDS staging overlapped with WMMAs.
// ---------------------------------------------------------------------------
enum { EPI_QKV = 0, EPI_RES = 1, EPI_GELU = 2 };

template <int EPI>
__global__ __launch_bounds__(256)
void gemm_wmma_bf16(const bf16* __restrict__ A, const bf16* __restrict__ Bt,
                    const float* __restrict__ bias, const float* __restrict__ resid,
                    bf16* __restrict__ outB, float* __restrict__ outF,
                    int M, int N, int K) {
  __shared__ bf16 As[2][128 * 40];   // [m][k], stride 40 (16B-aligned rows, padded)
  __shared__ bf16 Bs[2][128 * 40];   // [n][k], stride 40

  const int t    = threadIdx.x;
  const int lane = t & 31, wave = t >> 5;
  const int wm   = wave >> 2, wn = wave & 3;
  const int r16  = lane & 15, hsel = lane >> 4, kb = hsel * 8;
  const int bm   = blockIdx.y * 128, bn = blockIdx.x * 128;

  const int s_row = t >> 1, s_col = (t & 1) * 16;   // 128 rows x 32 cols, 16/thread

  f32x8 acc[4][2];
#pragma unroll
  for (int mi = 0; mi < 4; ++mi)
#pragma unroll
    for (int ni = 0; ni < 2; ++ni)
#pragma unroll
      for (int e = 0; e < 8; ++e) acc[mi][ni][e] = 0.0f;

  const bf16* ag = A  + (size_t)(bm + s_row) * K + s_col;
  const bf16* bg = Bt + (size_t)(bn + s_row) * K + s_col;
  bf16* const al = &As[0][0] + s_row * 40 + s_col;
  bf16* const bl = &Bs[0][0] + s_row * 40 + s_col;
  const int lds_buf = 128 * 40;

  auto stage = [&](int k0, int buf) {
    cp16B_g2l(ag + k0,     al + buf * lds_buf);
    cp16B_g2l(ag + k0 + 8, al + buf * lds_buf + 8);
    cp16B_g2l(bg + k0,     bl + buf * lds_buf);
    cp16B_g2l(bg + k0 + 8, bl + buf * lds_buf + 8);
  };

  stage(0, 0);
  int cur = 0;
  for (int k0 = 0; k0 < K; k0 += 32) {
    wait_async_all();          // staged tile (this iter's 'cur') is in LDS
    __syncthreads();           // ...and visible to all waves
    if (k0 + 32 < K) stage(k0 + 32, cur ^ 1);   // prefetch next tile via DMA

    FragAB af[4], bfr[2];
#pragma unroll
    for (int mi = 0; mi < 4; ++mi) {
      const bf16* p = &As[cur][(wm * 64 + mi * 16 + r16) * 40 + kb];
      af[mi].q[0] = *(const uint4*)(p);
      af[mi].q[1] = *(const uint4*)(p + 16);
    }
#pragma unroll
    for (int ni = 0; ni < 2; ++ni) {
      const bf16* p = &Bs[cur][(wn * 32 + ni * 16 + r16) * 40 + kb];
      bfr[ni].q[0] = *(const uint4*)(p);
      bfr[ni].q[1] = *(const uint4*)(p + 16);
    }
#pragma unroll
    for (int mi = 0; mi < 4; ++mi)
#pragma unroll
      for (int ni = 0; ni < 2; ++ni)
        acc[mi][ni] = __builtin_amdgcn_wmma_f32_16x16x32_bf16(
            false, af[mi].v, false, bfr[ni].v, (short)0, acc[mi][ni], false, false);
    cur ^= 1;
  }

  // Epilogue (C/D layout: VGPR r -> row r / r+8 by lane half; col = lane&15)
#pragma unroll
  for (int mi = 0; mi < 4; ++mi) {
#pragma unroll
    for (int ni = 0; ni < 2; ++ni) {
#pragma unroll
      for (int r = 0; r < 8; ++r) {
        const int row = bm + wm * 64 + mi * 16 + hsel * 8 + r;
        const int col = bn + wn * 32 + ni * 16 + r16;
        const size_t idx = (size_t)row * N + col;
        float v = acc[mi][ni][r] + bias[col];
        if (EPI == EPI_QKV)       outB[idx] = to_bf16(v);
        else if (EPI == EPI_RES)  outF[idx] = v + resid[idx];
        else                      outB[idx] = to_bf16(gelu_tanh(v));
      }
    }
  }
}

// ---------------------------------------------------------------------------
// Causal flash attention, hd=64, 16 heads, WMMA bf16.
// Block = 128 threads (4 waves), 64 q-rows per block (16 per wave).
// K and V^T tiles async-DMA staged into LDS; online softmax; PV via WMMA.
// ---------------------------------------------------------------------------
__global__ __launch_bounds__(128)
void attn_wmma_kernel(const bf16* __restrict__ qkv, const bf16* __restrict__ vt,
                      bf16* __restrict__ y, int S) {
  const int D = 1024, TD = 3072, HD = 64;
  const int b = blockIdx.z, h = blockIdx.y, qblk = blockIdx.x;
  const int t = threadIdx.x, lane = t & 31, wave = t >> 5;
  const int r16 = lane & 15, hsel = lane >> 4, kb = hsel * 8;
  const int q0 = qblk * 64 + wave * 16;

  __shared__ bf16 Ks[32 * 72];     // [key][hd], stride 72
  __shared__ bf16 Vt[64 * 40];     // [hd][key], stride 40
  __shared__ bf16 Ps[4][16 * 40];  // per-wave P tile [row][key], stride 40

  const bf16* base = qkv + (size_t)b * S * TD;
  const bf16* Qp  = base + h * HD;
  const bf16* Kp  = base + D + h * HD;
  const bf16* Vtp = vt + (size_t)(b * 16 + h) * HD * S;   // [hd][S]

  // Preload Q fragments (hd=64 -> 2 WMMA K-steps of 32)
  FragAB qf[2];
#pragma unroll
  for (int kk = 0; kk < 2; ++kk) {
    const bf16* p = Qp + (size_t)(q0 + r16) * TD + kk * 32 + kb;
    qf[kk].q[0] = *(const uint4*)(p);
    qf[kk].q[1] = *(const uint4*)(p + 16);
  }

  f32x8 o[4];
#pragma unroll
  for (int i = 0; i < 4; ++i)
#pragma unroll
    for (int e = 0; e < 8; ++e) o[i][e] = 0.0f;
  float mrow[8], lrow[8];
#pragma unroll
  for (int r = 0; r < 8; ++r) { mrow[r] = -__builtin_inff(); lrow[r] = 0.0f; }

  const int ld_key = t >> 2, ld_hc = (t & 3) * 16;   // K tile: 32 keys x 64 hd
  const int vd = t >> 1, vc = (t & 1) * 16;          // V^T tile: 64 hd x 32 keys
  const int nkt = qblk * 2 + 2;                      // causal: keys up to qblk*64+63
  for (int kt = 0; kt < nkt; ++kt) {
    // K tile: async DMA straight into LDS (row-major, no reshuffle needed)
    {
      const bf16* kg = Kp + (size_t)(kt * 32 + ld_key) * TD + ld_hc;
      cp16B_g2l(kg,     &Ks[ld_key * 72 + ld_hc]);
      cp16B_g2l(kg + 8, &Ks[ld_key * 72 + ld_hc + 8]);
    }
    // V^T tile: async DMA from pre-transposed vt (contiguous keys per hd row)
    {
      const bf16* vg = Vtp + (size_t)vd * S + kt * 32 + vc;
      cp16B_g2l(vg,     &Vt[vd * 40 + vc]);
      cp16B_g2l(vg + 8, &Vt[vd * 40 + vc + 8]);
    }
    wait_async_all();
    __syncthreads();

    if (kt * 32 <= q0 + 15) {       // wave-uniform -> EXEC stays all-ones for WMMA
      f32x8 sc[2];
#pragma unroll
      for (int nt = 0; nt < 2; ++nt)
#pragma unroll
        for (int e = 0; e < 8; ++e) sc[nt][e] = 0.0f;
#pragma unroll
      for (int kk = 0; kk < 2; ++kk) {
#pragma unroll
        for (int nt = 0; nt < 2; ++nt) {
          FragAB kf;
          const bf16* p = &Ks[(nt * 16 + r16) * 72 + kk * 32 + kb];
          kf.q[0] = *(const uint4*)(p);
          kf.q[1] = *(const uint4*)(p + 16);
          sc[nt] = __builtin_amdgcn_wmma_f32_16x16x32_bf16(
              false, qf[kk].v, false, kf.v, (short)0, sc[nt], false, false);
        }
      }
      // Online softmax; row reductions within 16-lane halves
#pragma unroll
      for (int r = 0; r < 8; ++r) {
        const int qrow = q0 + hsel * 8 + r;
        const int key0 = kt * 32 + r16, key1 = kt * 32 + 16 + r16;
        float v0 = sc[0][r] * 0.125f;     // hd^-0.5
        float v1 = sc[1][r] * 0.125f;
        if (key0 > qrow) v0 = -__builtin_inff();
        if (key1 > qrow) v1 = -__builtin_inff();
        float mx = fmaxf(v0, v1);
#pragma unroll
        for (int off = 8; off >= 1; off >>= 1) mx = fmaxf(mx, __shfl_xor(mx, off, 32));
        float mnew = fmaxf(mrow[r], mx);
        float p0, p1, alpha;
        if (mnew == -__builtin_inff()) { p0 = 0.f; p1 = 0.f; alpha = 1.f; }
        else {
          alpha = __expf(mrow[r] - mnew);
          p0 = __expf(v0 - mnew);
          p1 = __expf(v1 - mnew);
        }
        float rs = p0 + p1;
#pragma unroll
        for (int off = 8; off >= 1; off >>= 1) rs += __shfl_xor(rs, off, 32);
        lrow[r] = lrow[r] * alpha + rs;
        mrow[r] = mnew;
#pragma unroll
        for (int ni = 0; ni < 4; ++ni) o[ni][r] *= alpha;
        const int prow = hsel * 8 + r;
        Ps[wave][prow * 40 + r16]      = to_bf16(p0);
        Ps[wave][prow * 40 + 16 + r16] = to_bf16(p1);
      }
      // Re-layout P through LDS into A-fragment format, then PV WMMAs
      FragAB pf;
      const bf16* pp = &Ps[wave][r16 * 40 + kb];
      pf.q[0] = *(const uint4*)(pp);
      pf.q[1] = *(const uint4*)(pp + 16);
#pragma unroll
      for (int ni = 0; ni < 4; ++ni) {
        FragAB vf;
        const bf16* vp = &Vt[(ni * 16 + r16) * 40 + kb];
        vf.q[0] = *(const uint4*)(vp);
        vf.q[1] = *(const uint4*)(vp + 16);
        o[ni] = __builtin_amdgcn_wmma_f32_16x16x32_bf16(
            false, pf.v, false, vf.v, (short)0, o[ni], false, false);
      }
    }
    __syncthreads();
  }

  // Normalize and write y (bf16), layout [B*S, D]
#pragma unroll
  for (int r = 0; r < 8; ++r) {
    const float inv = 1.0f / lrow[r];
    const size_t row = (size_t)b * S + q0 + hsel * 8 + r;
#pragma unroll
    for (int ni = 0; ni < 4; ++ni)
      y[row * D + h * HD + ni * 16 + r16] = to_bf16(o[ni][r] * inv);
  }
}

// ---------------------------------------------------------------------------
// Launch: LN1 -> QKV GEMM -> V transpose -> attention -> proj(+res) -> LN2
//         -> FC(gelu) -> FCproj(+res)
// ---------------------------------------------------------------------------
extern "C" void kernel_launch(void* const* d_in, const int* in_sizes, int n_in,
                              void* d_out, int out_size, void* d_ws, size_t ws_size,
                              hipStream_t stream) {
  (void)in_sizes; (void)n_in; (void)out_size; (void)ws_size;
  const float* x      = (const float*)d_in[0];
  const float* ln1_g  = (const float*)d_in[1];
  const float* ln1_b  = (const float*)d_in[2];
  const float* ln2_g  = (const float*)d_in[3];
  const float* ln2_b  = (const float*)d_in[4];
  const float* W_attn = (const float*)d_in[5];
  const float* b_attn = (const float*)d_in[6];
  const float* W_proj = (const float*)d_in[7];
  const float* b_proj = (const float*)d_in[8];
  const float* W_fc   = (const float*)d_in[9];
  const float* b_fc   = (const float*)d_in[10];
  const float* W_fcp  = (const float*)d_in[11];
  const float* b_fcp  = (const float*)d_in[12];
  float* out = (float*)d_out;

  const int B = 2, S = 2048, D = 1024, M = B * S, H4 = 4 * D, TD = 3 * D;

  char* ws = (char*)d_ws;
  size_t off = 0;
  auto carve = [&](size_t bytes) -> void* {
    void* p = ws + off;
    off += (bytes + 255) & ~(size_t)255;
    return p;
  };
  // Weights stored TRANSPOSED in bf16: Wt[N][K]
  bf16*  Wattn_t = (bf16*)carve((size_t)D * TD * 2);
  bf16*  Wproj_t = (bf16*)carve((size_t)D * D * 2);
  bf16*  Wfc_t   = (bf16*)carve((size_t)D * H4 * 2);
  bf16*  Wfcp_t  = (bf16*)carve((size_t)H4 * D * 2);
  bf16*  h_b     = (bf16*)carve((size_t)M * D * 2);
  bf16*  qkv_b   = (bf16*)carve((size_t)M * TD * 2);
  bf16*  vt_b    = (bf16*)carve((size_t)M * D * 2);   // [B*H, 64, S]
  bf16*  y_b     = (bf16*)carve((size_t)M * D * 2);
  float* x1      = (float*)carve((size_t)M * D * 4);
  bf16*  h2_b    = (bf16*)carve((size_t)M * D * 2);
  bf16*  act_b   = (bf16*)carve((size_t)M * H4 * 2);

  // in: [K][N] fp32 -> out: [N][K] bf16
  auto cvtT = [&](const float* src, bf16* dst, int K, int N) {
    cvt_transpose_f32_bf16<<<dim3(N / 32, K / 32), 256, 0, stream>>>(src, dst, K, N);
  };
  cvtT(W_attn, Wattn_t, D,  TD);
  cvtT(W_proj, Wproj_t, D,  D);
  cvtT(W_fc,   Wfc_t,   D,  H4);
  cvtT(W_fcp,  Wfcp_t,  H4, D);

  ln_bf16_kernel<<<M, 256, 0, stream>>>(x, ln1_g, ln1_b, h_b);

  gemm_wmma_bf16<EPI_QKV><<<dim3(TD / 128, M / 128), 256, 0, stream>>>(
      h_b, Wattn_t, b_attn, nullptr, qkv_b, nullptr, M, TD, D);

  transpose_v_kernel<<<dim3(S / 32, 32, B), 256, 0, stream>>>(qkv_b, vt_b, S);

  attn_wmma_kernel<<<dim3(S / 64, 16, B), 128, 0, stream>>>(qkv_b, vt_b, y_b, S);

  gemm_wmma_bf16<EPI_RES><<<dim3(D / 128, M / 128), 256, 0, stream>>>(
      y_b, Wproj_t, b_proj, x, nullptr, x1, M, D, D);

  ln_bf16_kernel<<<M, 256, 0, stream>>>(x1, ln2_g, ln2_b, h2_b);

  gemm_wmma_bf16<EPI_GELU><<<dim3(H4 / 128, M / 128), 256, 0, stream>>>(
      h2_b, Wfc_t, b_fc, nullptr, act_b, nullptr, M, H4, D);

  gemm_wmma_bf16<EPI_RES><<<dim3(D / 128, M / 128), 256, 0, stream>>>(
      act_b, Wfcp_t, b_fcp, x1, nullptr, out, M, D, H4);
}
